// SPGAT_36867999268897
// MI455X (gfx1250) — matrix operand order repaired and checked by reading, not compiled, not run on hardware
//
#include <hip/hip_runtime.h>
#include <hip/hip_bf16.h>

// ---------------- problem constants ----------------
#define B_    8
#define T_    12
#define BT_   96
#define N_    256
#define OIN_  64
#define HID_  64
#define OUT_  64
#define HEADS_ 4

typedef __bf16 bf16_t;
typedef __attribute__((ext_vector_type(16))) __bf16 v16bf;
typedef __attribute__((ext_vector_type(8)))  float  v8f;

// ---------------- WMMA fragment loaders (wave32; contiguous k order, same in A and B) ----
// A (16x32 bf16, row-major in LDS): lane L(0-15) row M=L holds K 0..15,
// lane L+16 same row holds K 16..31; VGPR v packs a K pair.
// Per-lane memory pattern: one contiguous 32B run -> merges to ds_load_b128.
static __device__ __forceinline__ v16bf ld_a(const bf16_t* p, int row0, int k0, int ld) {
  const int ln = threadIdx.x & 31;
  const int m = ln & 15, half = ln >> 4;
  const bf16_t* base = p + (size_t)(row0 + m) * ld + k0 + half * 16;
  v16bf a;
#pragma unroll
  for (int v = 0; v < 8; ++v) {
    const int k = 2 * v;
    a[2 * v]     = base[k];
    a[2 * v + 1] = base[k + 1];
  }
  return a;
}

// B (32x16 bf16) from row-major [K][N] storage: lane L(0-15) col N=L holds
// K 0..15, lane L+16 same col holds K 16..31. Strided per lane (used only for
// loop-invariant weight tiles which the compiler hoists).
static __device__ __forceinline__ v16bf ld_b(const bf16_t* p, int k0, int col0, int ld) {
  const int ln = threadIdx.x & 31;
  const int c = ln & 15, half = ln >> 4;
  const bf16_t* base = p + (size_t)(k0 + half * 16) * ld + col0 + c;
  v16bf b;
#pragma unroll
  for (int v = 0; v < 8; ++v) {
    b[2 * v]     = base[(size_t)(2 * v) * ld];
    b[2 * v + 1] = base[(size_t)(2 * v + 1) * ld];
  }
  return b;
}

// B (32x16 bf16) from column-major [N][K] storage (element (k,c) at p[c*ldk+k]):
// per-lane 16 contiguous K values (32B run) -> merges to 2x ds_load_b128.
static __device__ __forceinline__ v16bf ld_bT(const bf16_t* p, int k0, int col0, int ldk) {
  const int ln = threadIdx.x & 31;
  const int c = ln & 15, half = ln >> 4;
  const bf16_t* base = p + (size_t)(col0 + c) * ldk + k0 + half * 16;
  v16bf b;
#pragma unroll
  for (int v = 0; v < 8; ++v) {
    b[2 * v]     = base[2 * v];
    b[2 * v + 1] = base[2 * v + 1];
  }
  return b;
}

static __device__ __forceinline__ v8f wmma_bf16(v16bf a, v16bf b, v8f c) {
  return __builtin_amdgcn_wmma_f32_16x16x32_bf16(false, a, false, b, (short)0, c, false, false);
}

// ---------------- kernel 1: tiny embedding chains -> c2[N][64], c3[BT][64] ----
// c2[n]  = ((concat(sfc,pfc)@Wout)+bout) @ Wout_bot
// c3[bt] = (tfc @ Wout_bot) + bout + b2 @ Wout_top   (b2 of GAT2 folded here)
__global__ void k_pre(const float* te, const float* se, const float* pe,
                      const float* Wt, const float* bt_b, const float* Ws, const float* bs,
                      const float* Wp, const float* bp, const float* Wfc, const float* bfc,
                      const float* Wout, const float* bout, const float* b2v,
                      float* c2, float* c3) {
  __shared__ float sh_a[64], sh_b[64], sh_c[64];
  const int tid = threadIdx.x;   // 64 threads
  const int blk = blockIdx.x;
  if (blk < N_) {
    const int n = blk;
    float acc = bs[tid];
    for (int k = 0; k < 32; ++k) acc += se[n * 32 + k] * Ws[k * 64 + tid];
    sh_a[tid] = acc;
    __syncthreads();
    float sfc = bfc[tid];
    for (int k = 0; k < 64; ++k) sfc += sh_a[k] * Wfc[k * 64 + tid];
    sh_b[tid] = sfc;
    __syncthreads();
    acc = bp[tid];
    for (int k = 0; k < 32; ++k) acc += pe[n * 32 + k] * Wp[k * 64 + tid];
    sh_a[tid] = acc;
    __syncthreads();
    float pfc = bfc[tid];
    for (int k = 0; k < 64; ++k) pfc += sh_a[k] * Wfc[k * 64 + tid];
    sh_c[tid] = pfc;
    __syncthreads();
    float sp = bout[tid];
    for (int k = 0; k < 64; ++k) sp += sh_b[k] * Wout[k * 64 + tid];
    for (int k = 0; k < 64; ++k) sp += sh_c[k] * Wout[(64 + k) * 64 + tid];
    __syncthreads();
    sh_a[tid] = sp;
    __syncthreads();
    float o = 0.f;
    for (int k = 0; k < 64; ++k) o += sh_a[k] * Wout[(64 + k) * 64 + tid];
    c2[n * 64 + tid] = o;
  } else {
    const int bt = blk - N_;
    float acc = bt_b[tid];
    for (int k = 0; k < 32; ++k) acc += te[bt * 32 + k] * Wt[k * 64 + tid];
    sh_a[tid] = acc;
    __syncthreads();
    float tfc = bfc[tid];
    for (int k = 0; k < 64; ++k) tfc += sh_a[k] * Wfc[k * 64 + tid];
    __syncthreads();
    sh_a[tid] = tfc;
    __syncthreads();
    float o = bout[tid];
    for (int k = 0; k < 64; ++k) o += sh_a[k] * Wout[(64 + k) * 64 + tid];
    for (int k = 0; k < 64; ++k) o += b2v[k] * Wout[k * 64 + tid];
    c3[bt * 64 + tid] = o;
  }
}

// ---------------- kernel 2a: hp1T[bt][feat][node] = (x[bt] @ W1)^T ------------
// x: 256x64, W1: 64x256; output stored feature-major for aggregation B-operand.
__global__ void __launch_bounds__(256) k_hp1(const float* x, const float* W1, bf16_t* hp1T) {
  __shared__ __align__(16) bf16_t xs[N_ * 64];    // 32 KB
  __shared__ __align__(16) bf16_t w1s[64 * 128];  // 16 KB (half of W1 cols)
  const int bt = blockIdx.x;
  const int tid = threadIdx.x;
  const int wave = tid >> 5;
  const float* xb = x + (size_t)bt * N_ * OIN_;
  for (int e = tid; e < N_ * 64; e += 256) xs[e] = (bf16_t)xb[e];
  bf16_t* out = hp1T + (size_t)bt * 256 * N_;
  for (int ch = 0; ch < 2; ++ch) {
    __syncthreads();
    for (int e = tid; e < 64 * 128; e += 256)
      w1s[e] = (bf16_t)W1[(e >> 7) * 256 + ch * 128 + (e & 127)];
    __syncthreads();
    // 16 mt x 8 nt tiles; wave w owns column strip nt=w (B hoisted), all mt
    for (int t = wave; t < 128; t += 8) {
      const int mt = t >> 3, nt = t & 7;
      v8f acc = {};
      acc = wmma_bf16(ld_a(xs, mt * 16, 0, 64),  ld_b(w1s, 0, nt * 16, 128), acc);
      acc = wmma_bf16(ld_a(xs, mt * 16, 32, 64), ld_b(w1s, 32, nt * 16, 128), acc);
      const int ln = tid & 31, c = ln & 15, half = ln >> 4;
      const int feat = ch * 128 + nt * 16 + c;
#pragma unroll
      for (int r = 0; r < 8; ++r)                  // consecutive nodes -> merges
        out[(size_t)feat * N_ + mt * 16 + half * 8 + r] = (bf16_t)acc[r];
    }
  }
}

// ---------------- kernel 2a2: per-head attention logits of GAT1 ---------------
// feature-major hp1T -> lane index == node == contiguous: fully coalesced.
__global__ void k_logits1(const bf16_t* hp1T, const float* a1_src, const float* a1_dst,
                          float* es1, float* ed1) {
  const int bt = blockIdx.x, i = threadIdx.x;
  const bf16_t* base = hp1T + (size_t)bt * 256 * N_;
  for (int h = 0; h < HEADS_; ++h) {
    float es = 0.f, ed = 0.f;
    for (int d = 0; d < 64; ++d) {
      const float v = (float)base[(size_t)(h * 64 + d) * N_ + i];
      es += v * a1_src[h * 64 + d];
      ed += v * a1_dst[h * 64 + d];
    }
    es1[((size_t)bt * HEADS_ + h) * N_ + i] = es;
    ed1[((size_t)bt * HEADS_ + h) * N_ + i] = ed;
  }
}

// ---------------- kernel 2b: GAT1 masked softmax + aggregate + bias + elu -----
// grid (BT, HEADS, 8 j-chunks of 32), 256 threads
__global__ void __launch_bounds__(256) k_gat1(const int* adj, const bf16_t* hp1T,
                                              const float* es1, const float* ed1,
                                              const float* b1, bf16_t* hout) {
  __shared__ __align__(16) bf16_t hp_sT[64 * N_];    // 32 KB: [feat d][node i]
  __shared__ __align__(16) bf16_t alpha_s[32 * N_];  // 16 KB: [j][i]
  __shared__ float es_s[N_], ed_s[32], red[8][32], colmax_s[32], colrcp[32];
  const int bt = blockIdx.x, h = blockIdx.y, jblk = blockIdx.z;
  const int tid = threadIdx.x;
  {  // stage head slice, feature-major: thread (d=tid>>2, quarter=tid&3)
    const int dr = tid >> 2, q = tid & 3;
    const uint4* src = (const uint4*)(hp1T + (size_t)bt * 256 * N_ +
                                      (size_t)(h * 64 + dr) * N_ + q * 64);
    uint4* dst = (uint4*)(hp_sT + dr * N_ + q * 64);
#pragma unroll
    for (int v = 0; v < 8; ++v) dst[v] = src[v];
  }
  es_s[tid] = es1[((size_t)bt * HEADS_ + h) * N_ + tid];
  if (tid < 32) ed_s[tid] = ed1[((size_t)bt * HEADS_ + h) * N_ + jblk * 32 + tid];
  __syncthreads();

  const int jj = tid & 31, slice = tid >> 5;
  const int jg = jblk * 32 + jj;
  const int* adjcol = adj + (size_t)bt * N_ * N_ + jg;   // + i*N_
  const float edj = ed_s[jj];
  float m = -3.0e38f;
  for (int i = slice * 32; i < slice * 32 + 32; ++i)
    if (adjcol[(size_t)i * N_] != 0) {
      float v = es_s[i] + edj; v = v > 0.f ? v : 0.2f * v;
      m = fmaxf(m, v);
    }
  red[slice][jj] = m;
  __syncthreads();
  if (slice == 0) {
    float M = red[0][jj];
#pragma unroll
    for (int s2 = 1; s2 < 8; ++s2) M = fmaxf(M, red[s2][jj]);
    colmax_s[jj] = M;
  }
  __syncthreads();
  const float M = colmax_s[jj];
  float sm = 0.f;
  for (int i = slice * 32; i < slice * 32 + 32; ++i) {   // contiguous i-run
    float e = 0.f;
    if (adjcol[(size_t)i * N_] != 0) {
      float v = es_s[i] + edj; v = v > 0.f ? v : 0.2f * v;
      e = __expf(v - M);
      sm += e;
    }
    alpha_s[jj * N_ + i] = (bf16_t)e;   // unnormalized; normalize post-WMMA
  }
  red[slice][jj] = sm;
  __syncthreads();
  if (slice == 0) {
    float S = 0.f;
#pragma unroll
    for (int s2 = 0; s2 < 8; ++s2) S += red[s2][jj];
    colrcp[jj] = S > 0.f ? 1.0f / S : 0.f;
  }
  __syncthreads();

  // agg tile [32 j][64 d] = 2x4 WMMA tiles, one per wave; K=256 -> 8 steps.
  // A = alpha_s (row-major [j][i]); B = hp_sT (col-major-in-K [d][i]).
  const int wave = tid >> 5;
  const int jt = wave >> 2, dt = wave & 3;
  v8f acc = {};
#pragma unroll
  for (int kk = 0; kk < 8; ++kk)
    acc = wmma_bf16(ld_a(alpha_s, jt * 16, kk * 32, N_),
                    ld_bT(hp_sT, kk * 32, dt * 16, N_), acc);
  const int ln = tid & 31, c = ln & 15, half = ln >> 4;
#pragma unroll
  for (int r = 0; r < 8; ++r) {
    const int jl = jt * 16 + half * 8 + r;
    const int d = dt * 16 + c;
    const float v = colrcp[jl] * acc[r] + b1[h * 64 + d];
    const float o = v > 0.f ? v : expm1f(v);  // elu
    hout[((size_t)bt * N_ + jblk * 32 + jl) * 256 + h * 64 + d] = (bf16_t)o;
  }
}

// ---------------- kernel 3a: hp2T[bt][feat][node] = (h[bt] @ W2)^T ------------
__global__ void __launch_bounds__(256) k_hp2(const bf16_t* hin, const float* W2, bf16_t* hp2T) {
  __shared__ __align__(16) bf16_t hs[N_ * 64];   // 32 KB K-chunk of h (row-major)
  __shared__ __align__(16) bf16_t w2s[64 * 64];  // 8 KB
  const int bt = blockIdx.x, tid = threadIdx.x;
  const int wave = tid >> 5;
  v8f acc[8];
#pragma unroll
  for (int t = 0; t < 8; ++t) acc[t] = (v8f){};
  for (int kc = 0; kc < 4; ++kc) {
    __syncthreads();
    {
      const uint4* src = (const uint4*)(hin + ((size_t)bt * N_ + tid) * 256 + kc * 64);
      uint4* dst = (uint4*)(hs + tid * 64);
#pragma unroll
      for (int v = 0; v < 8; ++v) dst[v] = src[v];
    }
    for (int e = tid; e < 64 * 64; e += 256)
      w2s[e] = (bf16_t)W2[(kc * 64 + (e >> 6)) * 64 + (e & 63)];
    __syncthreads();
#pragma unroll
    for (int t = 0; t < 8; ++t) {
      const int tile = wave + t * 8;          // 16 mt x 4 nt; nt fixed per wave
      const int mt = tile >> 2, nt = tile & 3;
      acc[t] = wmma_bf16(ld_a(hs, mt * 16, 0, 64),  ld_b(w2s, 0, nt * 16, 64), acc[t]);
      acc[t] = wmma_bf16(ld_a(hs, mt * 16, 32, 64), ld_b(w2s, 32, nt * 16, 64), acc[t]);
    }
  }
  const int ln = tid & 31, c = ln & 15, half = ln >> 4;
  bf16_t* out = hp2T + (size_t)bt * 64 * N_;
#pragma unroll
  for (int t = 0; t < 8; ++t) {
    const int tile = wave + t * 8;
    const int mt = tile >> 2, nt = tile & 3;
#pragma unroll
    for (int r = 0; r < 8; ++r)                  // consecutive nodes -> merges
      out[(size_t)(nt * 16 + c) * N_ + mt * 16 + half * 8 + r] = (bf16_t)acc[t][r];
  }
}

// ---------------- kernel 3a2: attention logits of GAT2 ----------------
__global__ void k_logits2(const bf16_t* hp2T, const float* a2_src, const float* a2_dst,
                          float* es2, float* ed2) {
  const int bt = blockIdx.x, i = threadIdx.x;
  const bf16_t* base = hp2T + (size_t)bt * 64 * N_;
  float es = 0.f, ed = 0.f;
  for (int d = 0; d < 64; ++d) {
    const float v = (float)base[(size_t)d * N_ + i];
    es += v * a2_src[d];
    ed += v * a2_dst[d];
  }
  es2[(size_t)bt * N_ + i] = es;
  ed2[(size_t)bt * N_ + i] = ed;
}

// ---------------- kernel 3b: GAT2 softmax+aggregate, fused final GEMM ---------
// out = od @ Wout_top + c2[n] + c3[bt]   (b2 already folded into c3)
// grid (BT, 8 j-chunks of 32), 256 threads
__global__ void __launch_bounds__(256) k_gat2(const int* adj, const bf16_t* hp2T,
                                              const float* es2, const float* ed2,
                                              const float* Wout, const float* c2,
                                              const float* c3, float* outp) {
  __shared__ __align__(16) bf16_t hp_sT[64 * N_];    // 32 KB: [feat d][node i]
  __shared__ __align__(16) bf16_t alpha_s[32 * N_];  // 16 KB: [j][i]
  __shared__ __align__(16) bf16_t od_s[32 * 64];     // 4 KB (row-major, A operand)
  __shared__ __align__(16) bf16_t woutT_s[64 * 64];  // 8 KB: [o][k] (Wout_top^T)
  __shared__ float es_s[N_], ed_s[32], red[8][32], colmax_s[32], colrcp[32];
  const int bt = blockIdx.x, jblk = blockIdx.y, tid = threadIdx.x;
  {
    const int dr = tid >> 2, q = tid & 3;
    const uint4* src = (const uint4*)(hp2T + (size_t)bt * 64 * N_ +
                                      (size_t)dr * N_ + q * 64);
    uint4* dst = (uint4*)(hp_sT + dr * N_ + q * 64);
#pragma unroll
    for (int v = 0; v < 8; ++v) dst[v] = src[v];
  }
  for (int e = tid; e < 64 * 64; e += 256)      // coalesced read, transposed write
    woutT_s[(e & 63) * 64 + (e >> 6)] = (bf16_t)Wout[e];
  es_s[tid] = es2[(size_t)bt * N_ + tid];
  if (tid < 32) ed_s[tid] = ed2[(size_t)bt * N_ + jblk * 32 + tid];
  __syncthreads();

  const int jj = tid & 31, slice = tid >> 5;
  const int jg = jblk * 32 + jj;
  const int* adjcol = adj + (size_t)bt * N_ * N_ + jg;
  const float edj = ed_s[jj];
  float m = -3.0e38f;
  for (int i = slice * 32; i < slice * 32 + 32; ++i)
    if (adjcol[(size_t)i * N_] != 0) {
      float v = es_s[i] + edj; v = v > 0.f ? v : 0.2f * v;
      m = fmaxf(m, v);
    }
  red[slice][jj] = m;
  __syncthreads();
  if (slice == 0) {
    float M = red[0][jj];
#pragma unroll
    for (int s2 = 1; s2 < 8; ++s2) M = fmaxf(M, red[s2][jj]);
    colmax_s[jj] = M;
  }
  __syncthreads();
  const float M = colmax_s[jj];
  float sm = 0.f;
  for (int i = slice * 32; i < slice * 32 + 32; ++i) {
    float e = 0.f;
    if (adjcol[(size_t)i * N_] != 0) {
      float v = es_s[i] + edj; v = v > 0.f ? v : 0.2f * v;
      e = __expf(v - M);
      sm += e;
    }
    alpha_s[jj * N_ + i] = (bf16_t)e;
  }
  red[slice][jj] = sm;
  __syncthreads();
  if (slice == 0) {
    float S = 0.f;
#pragma unroll
    for (int s2 = 0; s2 < 8; ++s2) S += red[s2][jj];
    colrcp[jj] = S > 0.f ? 1.0f / S : 0.f;
  }
  __syncthreads();

  const int wave = tid >> 5;
  const int jt = wave >> 2, dt = wave & 3;
  const int ln = tid & 31, c = ln & 15, half = ln >> 4;
  // aggregate: od[32 j][64 d]
  {
    v8f acc = {};
#pragma unroll
    for (int kk = 0; kk < 8; ++kk)
      acc = wmma_bf16(ld_a(alpha_s, jt * 16, kk * 32, N_),
                      ld_bT(hp_sT, kk * 32, dt * 16, N_), acc);
#pragma unroll
    for (int r = 0; r < 8; ++r) {
      const int jl = jt * 16 + half * 8 + r;
      od_s[jl * 64 + dt * 16 + c] = (bf16_t)(colrcp[jl] * acc[r]);
    }
  }
  __syncthreads();
  // final: out[32 j][64 o] = od @ Wout_top + c2 + c3
  {
    v8f acc = {};
    acc = wmma_bf16(ld_a(od_s, jt * 16, 0, 64),  ld_bT(woutT_s, 0, dt * 16, 64), acc);
    acc = wmma_bf16(ld_a(od_s, jt * 16, 32, 64), ld_bT(woutT_s, 32, dt * 16, 64), acc);
#pragma unroll
    for (int r = 0; r < 8; ++r) {
      const int jl = jt * 16 + half * 8 + r;
      const int o = dt * 16 + c;
      const int n = jblk * 32 + jl;
      outp[((size_t)bt * N_ + n) * 64 + o] = acc[r] + c2[n * 64 + o] + c3[bt * 64 + o];
    }
  }
}

// ---------------- host-side launcher ----------------
extern "C" void kernel_launch(void* const* d_in, const int* in_sizes, int n_in,
                              void* d_out, int out_size, void* d_ws, size_t ws_size,
                              hipStream_t stream) {
  (void)in_sizes; (void)n_in; (void)out_size; (void)ws_size;
  const float* x      = (const float*)d_in[0];
  const int*   adj    = (const int*)  d_in[1];
  const float* te     = (const float*)d_in[2];
  const float* se     = (const float*)d_in[3];
  const float* pe     = (const float*)d_in[4];
  const float* Wt     = (const float*)d_in[5];
  const float* bt_b   = (const float*)d_in[6];
  const float* Ws     = (const float*)d_in[7];
  const float* bs     = (const float*)d_in[8];
  const float* Wp     = (const float*)d_in[9];
  const float* bp     = (const float*)d_in[10];
  const float* Wfc    = (const float*)d_in[11];
  const float* bfc    = (const float*)d_in[12];
  const float* Wout   = (const float*)d_in[13];
  const float* bout   = (const float*)d_in[14];
  const float* W1     = (const float*)d_in[15];
  const float* b1     = (const float*)d_in[16];
  const float* a1_src = (const float*)d_in[17];
  const float* a1_dst = (const float*)d_in[18];
  const float* W2     = (const float*)d_in[19];
  const float* b2     = (const float*)d_in[20];
  const float* a2_src = (const float*)d_in[21];
  const float* a2_dst = (const float*)d_in[22];
  float* outp = (float*)d_out;

  // workspace layout
  char* ws = (char*)d_ws;
  size_t off = 0;
  bf16_t* hp1T = (bf16_t*)(ws + off); off += (size_t)BT_ * 256 * N_ * 2;  // 12.6 MB
  bf16_t* hout = (bf16_t*)(ws + off); off += (size_t)BT_ * N_ * 256 * 2;  // 12.6 MB
  bf16_t* hp2T = (bf16_t*)(ws + off); off += (size_t)BT_ * 64 * N_ * 2;   // 3.1 MB
  float*  es1  = (float*) (ws + off); off += (size_t)BT_ * HEADS_ * N_ * 4;
  float*  ed1  = (float*) (ws + off); off += (size_t)BT_ * HEADS_ * N_ * 4;
  float*  es2  = (float*) (ws + off); off += (size_t)BT_ * N_ * 4;
  float*  ed2  = (float*) (ws + off); off += (size_t)BT_ * N_ * 4;
  float*  c2   = (float*) (ws + off); off += (size_t)N_ * 64 * 4;
  float*  c3   = (float*) (ws + off); off += (size_t)BT_ * 64 * 4;

  k_pre<<<dim3(N_ + BT_), dim3(64), 0, stream>>>(te, se, pe, Wt, bt_b, Ws, bs, Wp, bp,
                                                 Wfc, bfc, Wout, bout, b2, c2, c3);
  k_hp1<<<dim3(BT_), dim3(256), 0, stream>>>(x, W1, hp1T);
  k_logits1<<<dim3(BT_), dim3(N_), 0, stream>>>(hp1T, a1_src, a1_dst, es1, ed1);
  k_gat1<<<dim3(BT_, HEADS_, 8), dim3(256), 0, stream>>>(adj, hp1T, es1, ed1, b1, hout);
  k_hp2<<<dim3(BT_), dim3(256), 0, stream>>>(hout, W2, hp2T);
  k_logits2<<<dim3(BT_), dim3(N_), 0, stream>>>(hp2T, a2_src, a2_dst, es2, ed2);
  k_gat2<<<dim3(BT_, 8), dim3(256), 0, stream>>>(adj, hp2T, es2, ed2, Wout, c2, c3, outp);
}
